// multihead_net_12120397709660
// MI455X (gfx1250) — matrix operand order repaired and checked by reading, not compiled, not run on hardware
//
#include <hip/hip_runtime.h>
#include <stdint.h>

// Problem dims (fixed by the reference)
#define BB     32
#define TT     512
#define DD     512
#define H1     1024
#define HDIM   2048
#define GDIM   6144        // 3*H
#define PP     1024
#define MM     (BB * TT)   // 16384
#define NTASKS 8
#define NPRIM  3
#define SPLITK 8

typedef __bf16 v16bf __attribute__((ext_vector_type(16)));
typedef float  v8f   __attribute__((ext_vector_type(8)));

union FragAB { v16bf v; uint4 u[2]; };
struct Frags { FragAB a0, a1, b0, b1, b2, b3; };

__device__ __forceinline__ uint16_t f2bf(float f) {
    uint32_t u = __float_as_uint(f);
    uint32_t r = (u + 0x7FFFu + ((u >> 16) & 1u)) >> 16;   // round-to-nearest-even
    return (uint16_t)r;
}

// A fragment: lanes 0-15 hold K[0..15]; lanes 16-31 K[16..31] (contiguous, matches B)
__device__ __forceinline__ void load_a(FragAB& f, const uint16_t* arow, int k0) {
    const uint4* p = (const uint4*)(arow + k0);
    f.u[0] = p[0];          // K: k0+off    .. +7
    f.u[1] = p[1];          // K: k0+off+8  .. +15
}
// B = W^T fragment: lane n (mod 16) holds 16 contiguous K of W row n; hi half gets K+16
__device__ __forceinline__ void load_b(FragAB& f, const uint16_t* wrow, int k0) {
    const uint4* p = (const uint4*)(wrow + k0);
    f.u[0] = p[0];
    f.u[1] = p[1];
}

// ---------------------------------------------------------------------------
// fp32 -> bf16 conversion
// ---------------------------------------------------------------------------
__global__ __launch_bounds__(256) void cvt_f32_bf16(const float* __restrict__ in,
                                                    uint16_t* __restrict__ out, int n) {
    int i = blockIdx.x * 256 + threadIdx.x;
    if (i < n) out[i] = f2bf(in[i]);
}

// ---------------------------------------------------------------------------
// C[M,N] = A[M,K](bf16) @ W[N,K](bf16)^T  (+bias)(+relu)
// Wave tile 32x64 (2 A-frags x 4 B-frags -> 8 WMMA/k-step), block tile 32x512.
// Double-buffered fragment pipeline: next k-step's loads issued before the
// WMMAs consuming the current ones, so waits overlap matrix math.
// grid.z = split-K chunks; chunk z writes Cf + z*M*N (deterministic partials).
// Requires: M % 32 == 0, N % 512 == 0, (K/gridDim.z) % 64 == 0.
// ---------------------------------------------------------------------------
__global__ __launch_bounds__(256) void gemm_bf16_wmma(
    const uint16_t* __restrict__ A,     // [M,K] bf16 row-major
    const uint16_t* __restrict__ W,     // [N,K] bf16 row-major
    const float*    __restrict__ bias,  // [N] or nullptr
    float*          __restrict__ Cf,    // [gridDim.z, M, N] fp32, or nullptr
    uint16_t*       __restrict__ Cbf,   // [M, N] bf16, or nullptr
    int M, int N, int K, int relu)
{
    const int lane = threadIdx.x & 31;
    const int wave = threadIdx.x >> 5;
    const int l15  = lane & 15;
    const int hi   = lane >> 4;                    // 0: lanes 0-15, 1: lanes 16-31
    const int m0 = blockIdx.y * 32;
    const int n0 = blockIdx.x * 512 + wave * 64;

    const int kc   = K / gridDim.z;
    const int kbeg = blockIdx.z * kc;
    const int kend = kbeg + kc;

    const uint16_t* arow0 = A + (size_t)(m0 + l15) * K + (hi ? 16 : 0);
    const uint16_t* arow1 = arow0 + (size_t)16 * K;
    const uint16_t* wrow0 = W + (size_t)(n0 + l15) * K + (hi ? 16 : 0);
    const uint16_t* wrow1 = wrow0 + (size_t)16 * K;
    const uint16_t* wrow2 = wrow0 + (size_t)32 * K;
    const uint16_t* wrow3 = wrow0 + (size_t)48 * K;

    v8f acc[2][4] = {};

    auto load_frags = [&](Frags& f, int k) {
        load_a(f.a0, arow0, k);
        load_a(f.a1, arow1, k);
        load_b(f.b0, wrow0, k);
        load_b(f.b1, wrow1, k);
        load_b(f.b2, wrow2, k);
        load_b(f.b3, wrow3, k);
    };
    auto mma8 = [&](const Frags& f) {
        acc[0][0] = __builtin_amdgcn_wmma_f32_16x16x32_bf16(false, f.a0.v, false, f.b0.v, (short)0, acc[0][0], false, false);
        acc[0][1] = __builtin_amdgcn_wmma_f32_16x16x32_bf16(false, f.a0.v, false, f.b1.v, (short)0, acc[0][1], false, false);
        acc[0][2] = __builtin_amdgcn_wmma_f32_16x16x32_bf16(false, f.a0.v, false, f.b2.v, (short)0, acc[0][2], false, false);
        acc[0][3] = __builtin_amdgcn_wmma_f32_16x16x32_bf16(false, f.a0.v, false, f.b3.v, (short)0, acc[0][3], false, false);
        acc[1][0] = __builtin_amdgcn_wmma_f32_16x16x32_bf16(false, f.a1.v, false, f.b0.v, (short)0, acc[1][0], false, false);
        acc[1][1] = __builtin_amdgcn_wmma_f32_16x16x32_bf16(false, f.a1.v, false, f.b1.v, (short)0, acc[1][1], false, false);
        acc[1][2] = __builtin_amdgcn_wmma_f32_16x16x32_bf16(false, f.a1.v, false, f.b2.v, (short)0, acc[1][2], false, false);
        acc[1][3] = __builtin_amdgcn_wmma_f32_16x16x32_bf16(false, f.a1.v, false, f.b3.v, (short)0, acc[1][3], false, false);
    };

    // Double-buffered pipeline over k (kc is a multiple of 64, so steps are even)
    Frags fA, fB;
    load_frags(fA, kbeg);
    int k0 = kbeg;
    for (; k0 + 64 < kend; k0 += 64) {
        load_frags(fB, k0 + 32);
        mma8(fA);
        load_frags(fA, k0 + 64);
        mma8(fB);
    }
    load_frags(fB, k0 + 32);
    mma8(fA);
    mma8(fB);

    float* CfZ = Cf ? (Cf + (size_t)blockIdx.z * (size_t)M * (size_t)N) : nullptr;
#pragma unroll
    for (int i = 0; i < 2; ++i) {
#pragma unroll
        for (int j = 0; j < 4; ++j) {
            const int ncol = n0 + j * 16 + l15;
            const float bv = (bias && blockIdx.z == 0) ? bias[ncol] : 0.0f;
#pragma unroll
            for (int r = 0; r < 8; ++r) {
                const int mrow = m0 + i * 16 + r + (hi ? 8 : 0);
                float v = acc[i][j][r] + bv;
                if (relu) v = fmaxf(v, 0.0f);
                const size_t o = (size_t)mrow * N + ncol;
                if (CfZ) CfZ[o] = v;
                if (Cbf) Cbf[o] = f2bf(v);
            }
        }
    }
}

// ---------------------------------------------------------------------------
// Fused GRU gates: sums SPLITK gh partials + biases, applies sigmoid/tanh,
// produces h_next (fp32 + bf16) and the bf16 output sequence slice hs[:,t,:].
// ---------------------------------------------------------------------------
__global__ __launch_bounds__(256) void gru_gates(
    const float* __restrict__ gi,       // [B*T, 3H]
    const float* __restrict__ ghp,      // [SPLITK, B, 3H] partials
    const float* __restrict__ b_hh,     // [3H]
    const float* __restrict__ h_prev,   // [B, H]
    float*       __restrict__ h_next,   // [B, H]
    uint16_t*    __restrict__ h_next_bf,// [B, H]
    uint16_t*    __restrict__ hs_bf,    // [B, T, H]
    int t)
{
    const int idx = blockIdx.x * 256 + threadIdx.x;   // [0, B*H)
    const int b   = idx >> 11;                        // / H (2048)
    const int hh  = idx & (HDIM - 1);

    const float* gir = gi + ((size_t)b * TT + t) * GDIM;
    const float i_r = gir[hh];
    const float i_z = gir[HDIM + hh];
    const float i_n = gir[2 * HDIM + hh];

    float h_r = b_hh[hh], h_z = b_hh[HDIM + hh], h_n = b_hh[2 * HDIM + hh];
#pragma unroll
    for (int z = 0; z < SPLITK; ++z) {
        const float* g = ghp + ((size_t)z * BB + b) * GDIM;
        h_r += g[hh];
        h_z += g[HDIM + hh];
        h_n += g[2 * HDIM + hh];
    }

    const float r  = 1.0f / (1.0f + __expf(-(i_r + h_r)));
    const float zz = 1.0f / (1.0f + __expf(-(i_z + h_z)));
    const float n  = tanhf(i_n + r * h_n);
    const float hp = h_prev[idx];
    const float hn = (1.0f - zz) * n + zz * hp;

    h_next[idx]    = hn;
    const uint16_t hb = f2bf(hn);
    h_next_bf[idx] = hb;
    hs_bf[((size_t)b * TT + t) * HDIM + hh] = hb;
}

// ---------------------------------------------------------------------------
// Per-task head: out[b,t,j] = dot(out_s[b,t,:], Wp[j, label[b], :]) + bp[j, label[b]]
// One wave per (b,t) row, float4 loads, wave-shuffle reduction.
// ---------------------------------------------------------------------------
__global__ __launch_bounds__(256) void head_kernel(
    const float* __restrict__ outs,   // [B*T, P]
    const float* __restrict__ Wp,     // [3, NTASKS, P]
    const float* __restrict__ bp,     // [3, NTASKS]
    const int*   __restrict__ labels, // [B]
    float*       __restrict__ out)    // [B*T, 3]
{
    const int row  = blockIdx.x * 8 + (threadIdx.x >> 5);
    const int lane = threadIdx.x & 31;
    const int b    = row >> 9;                      // / T (512)
    const int lab  = labels[b];

    const float* xr = outs + (size_t)row * PP;
    const float* w0 = Wp + ((size_t)0 * NTASKS + lab) * PP;
    const float* w1 = Wp + ((size_t)1 * NTASKS + lab) * PP;
    const float* w2 = Wp + ((size_t)2 * NTASKS + lab) * PP;

    float a0 = 0.f, a1 = 0.f, a2 = 0.f;
    for (int d = lane * 4; d < PP; d += 32 * 4) {
        const float4 xv = *(const float4*)(xr + d);
        const float4 v0 = *(const float4*)(w0 + d);
        const float4 v1 = *(const float4*)(w1 + d);
        const float4 v2 = *(const float4*)(w2 + d);
        a0 += xv.x * v0.x + xv.y * v0.y + xv.z * v0.z + xv.w * v0.w;
        a1 += xv.x * v1.x + xv.y * v1.y + xv.z * v1.z + xv.w * v1.w;
        a2 += xv.x * v2.x + xv.y * v2.y + xv.z * v2.z + xv.w * v2.w;
    }
#pragma unroll
    for (int off = 16; off > 0; off >>= 1) {
        a0 += __shfl_xor(a0, off, 32);
        a1 += __shfl_xor(a1, off, 32);
        a2 += __shfl_xor(a2, off, 32);
    }
    if (lane == 0) {
        out[(size_t)row * 3 + 0] = a0 + bp[0 * NTASKS + lab];
        out[(size_t)row * 3 + 1] = a1 + bp[1 * NTASKS + lab];
        out[(size_t)row * 3 + 2] = a2 + bp[2 * NTASKS + lab];
    }
}

// ---------------------------------------------------------------------------
extern "C" void kernel_launch(void* const* d_in, const int* in_sizes, int n_in,
                              void* d_out, int out_size, void* d_ws, size_t ws_size,
                              hipStream_t stream) {
    (void)in_sizes; (void)n_in; (void)out_size; (void)ws_size;

    const float* x    = (const float*)d_in[0];
    const int*   tlab = (const int*)  d_in[1];
    const float* h0   = (const float*)d_in[2];
    const float* W1   = (const float*)d_in[3];
    const float* b1   = (const float*)d_in[4];
    const float* W2   = (const float*)d_in[5];
    const float* b2   = (const float*)d_in[6];
    const float* W_ih = (const float*)d_in[7];
    const float* b_ih = (const float*)d_in[8];
    const float* W_hh = (const float*)d_in[9];
    const float* b_hh = (const float*)d_in[10];
    const float* W3   = (const float*)d_in[11];
    const float* b3   = (const float*)d_in[12];
    const float* Wp   = (const float*)d_in[13];
    const float* bp   = (const float*)d_in[14];
    float* out = (float*)d_out;

    char* ws = (char*)d_ws;
    size_t off = 0;
    auto alloc = [&](size_t bytes) -> char* {
        char* p = ws + off;
        off += (bytes + 255) & ~(size_t)255;
        return p;
    };

    uint16_t* x_bf   = (uint16_t*)alloc((size_t)MM * DD * 2);
    uint16_t* W1_bf  = (uint16_t*)alloc((size_t)H1 * DD * 2);
    uint16_t* W2_bf  = (uint16_t*)alloc((size_t)HDIM * H1 * 2);
    uint16_t* Wih_bf = (uint16_t*)alloc((size_t)GDIM * HDIM * 2);
    uint16_t* Whh_bf = (uint16_t*)alloc((size_t)GDIM * HDIM * 2);
    uint16_t* W3_bf  = (uint16_t*)alloc((size_t)PP * HDIM * 2);
    uint16_t* a1_bf  = (uint16_t*)alloc((size_t)MM * H1 * 2);
    uint16_t* a2_bf  = (uint16_t*)alloc((size_t)MM * HDIM * 2);
    float*    gi_f   = (float*)   alloc((size_t)MM * GDIM * 4);
    uint16_t* hs_bf  = (uint16_t*)alloc((size_t)MM * HDIM * 2);
    float*    os_f   = (float*)   alloc((size_t)MM * PP * 4);
    float*    gh_f   = (float*)   alloc((size_t)SPLITK * BB * GDIM * 4);
    float*    hA_f   = (float*)   alloc((size_t)BB * HDIM * 4);
    float*    hB_f   = (float*)   alloc((size_t)BB * HDIM * 4);
    uint16_t* hA_bf  = (uint16_t*)alloc((size_t)BB * HDIM * 2);
    uint16_t* hB_bf  = (uint16_t*)alloc((size_t)BB * HDIM * 2);

    auto cvt = [&](const float* i, uint16_t* o, int n) {
        cvt_f32_bf16<<<(n + 255) / 256, 256, 0, stream>>>(i, o, n);
    };
    cvt(x,    x_bf,   MM * DD);
    cvt(W1,   W1_bf,  H1 * DD);
    cvt(W2,   W2_bf,  HDIM * H1);
    cvt(W_ih, Wih_bf, GDIM * HDIM);
    cvt(W_hh, Whh_bf, GDIM * HDIM);
    cvt(W3,   W3_bf,  PP * HDIM);
    hipMemcpyAsync(hA_f, h0, (size_t)BB * HDIM * 4, hipMemcpyDeviceToDevice, stream);
    cvt(h0, hA_bf, BB * HDIM);

    // L1: relu(x @ W1^T + b1) -> a1 (bf16)
    gemm_bf16_wmma<<<dim3(H1 / 512, MM / 32, 1), 256, 0, stream>>>(
        x_bf, W1_bf, b1, nullptr, a1_bf, MM, H1, DD, 1);
    // L2: relu(a1 @ W2^T + b2) -> a2 (bf16)
    gemm_bf16_wmma<<<dim3(HDIM / 512, MM / 32, 1), 256, 0, stream>>>(
        a1_bf, W2_bf, b2, nullptr, a2_bf, MM, HDIM, H1, 1);
    // gi = a2 @ W_ih^T + b_ih (fp32, one big GEMM instead of per-step)
    gemm_bf16_wmma<<<dim3(GDIM / 512, MM / 32, 1), 256, 0, stream>>>(
        a2_bf, Wih_bf, b_ih, gi_f, nullptr, MM, GDIM, HDIM, 0);

    // GRU recurrence: per step, split-K gh GEMM (W_hh is L2-resident) + fused gates
    float*    hprev_f  = hA_f;  uint16_t* hprev_bf = hA_bf;
    float*    hnext_f  = hB_f;  uint16_t* hnext_bf = hB_bf;
    for (int t = 0; t < TT; ++t) {
        gemm_bf16_wmma<<<dim3(GDIM / 512, 1, SPLITK), 256, 0, stream>>>(
            hprev_bf, Whh_bf, nullptr, gh_f, nullptr, BB, GDIM, HDIM, 0);
        gru_gates<<<(BB * HDIM) / 256, 256, 0, stream>>>(
            gi_f, gh_f, b_hh, hprev_f, hnext_f, hnext_bf, hs_bf, t);
        float* tf = hprev_f;     hprev_f = hnext_f;   hnext_f = tf;
        uint16_t* tb = hprev_bf; hprev_bf = hnext_bf; hnext_bf = tb;
    }

    // L3: relu(hs @ W3^T + b3) -> out_s (fp32)
    gemm_bf16_wmma<<<dim3(PP / 512, MM / 32, 1), 256, 0, stream>>>(
        hs_bf, W3_bf, b3, os_f, nullptr, MM, PP, HDIM, 1);

    // Per-task head -> out_primary
    head_kernel<<<MM / 8, 256, 0, stream>>>(os_f, Wp, bp, tlab, out);

    // h_last (out_aux is empty)
    hipMemcpyAsync(out + (size_t)MM * NPRIM, hprev_f,
                   (size_t)BB * HDIM * 4, hipMemcpyDeviceToDevice, stream);
}
